// StreamingClusterCompactor_14516989461216
// MI455X (gfx1250) — compile-verified, matrix operand order checked by reading, not run on hardware
//
#include <hip/hip_runtime.h>
#include <hip/hip_bf16.h>
#include <stdint.h>

typedef __bf16 v16bf __attribute__((ext_vector_type(16)));
typedef float  v8f   __attribute__((ext_vector_type(8)));

#define T_TOTAL 32768
#define H_TOTAL 32
#define D_DIM   128
#define HPG     4
#define G_GRP   (H_TOTAL / HPG)   // 8
#define M_ANCH  512
#define MT_CNT  (M_ANCH / 16)     // 32 m-tiles
#define FRAG_SZ (4 * 32 * 16)     // bf16 elems per m-tile (4 chunks x 32 lanes x 16)
#define WAVES_PER_WG 4
#define ROWS_PER_WAVE 16

__device__ __forceinline__ __bf16 f32_to_bf16_rne(float f) {
    uint32_t u = __float_as_uint(f);
    u += 0x7fffu + ((u >> 16) & 1u);           // round-to-nearest-even
    uint16_t hi = (uint16_t)(u >> 16);
    return __builtin_bit_cast(__bf16, hi);
}

// Monotonic float -> uint32 mapping (preserves total order of finite floats).
__device__ __forceinline__ uint32_t f32_orderable(float f) {
    uint32_t u = __float_as_uint(f);
    return u ^ (uint32_t)(((int32_t)u >> 31) | 0x80000000u);
}

// 16-bit WMMA operand layout (ISA 7.12.2): element j of the per-lane v16bf
// holds K index  k = j + (j<8?0:8) + (lane<16?0:8)  within a K=32 chunk.
__device__ __forceinline__ int frag_k(int j, int lane) {
    return j + ((j < 8) ? 0 : 8) + ((lane < 16) ? 0 : 8);
}

// ---------------------------------------------------------------- init ws --
__global__ void k_init(float* __restrict__ ws, int n) {
    int i = blockIdx.x * blockDim.x + threadIdx.x;
    if (i < n) ws[i] = 0.0f;
}

// ---- one-shot: convert anchors f32 -> bf16, pre-packed as WMMA A-frags ----
// Layout: wsA[g][mt][c][lane][j]  (16 bf16 = 32 contiguous bytes per lane)
__global__ void k_pack_anchors(const float* __restrict__ anchors,
                               __bf16* __restrict__ wsA)
{
    int tid = blockIdx.x * blockDim.x + threadIdx.x;   // one (g,mt,c,lane)
    if (tid >= G_GRP * MT_CNT * 4 * 32) return;
    int lane = tid & 31;
    int c    = (tid >> 5) & 3;
    int mt   = (tid >> 7) & (MT_CNT - 1);
    int g    = tid >> 12;

    int m = mt * 16 + (lane & 15);
    const float* src = anchors + ((size_t)g * M_ANCH + m) * D_DIM + c * 32;
    v16bf frag;
    #pragma unroll
    for (int j = 0; j < 16; ++j)
        frag[j] = f32_to_bf16_rne(src[frag_k(j, lane)]);
    *(v16bf*)(wsA + (size_t)tid * 16) = frag;
}

// ------------------------------------------------ route (WMMA) + scatter ---
__global__ __launch_bounds__(WAVES_PER_WG * 32)
void k_route_accum(const float* __restrict__ Kc, const float* __restrict__ Vc,
                   const __bf16* __restrict__ wsA,
                   float* __restrict__ wsK, float* __restrict__ wsV,
                   float* __restrict__ wsC)
{
    __shared__ float kg[WAVES_PER_WG][ROWS_PER_WAVE][D_DIM];   // 32 KB
    __shared__ float vg[WAVES_PER_WG][ROWS_PER_WAVE][D_DIM];   // 32 KB

    const int w    = threadIdx.x >> 5;
    const int lane = threadIdx.x & 31;
    const int g    = blockIdx.y;
    const int t0   = (blockIdx.x * WAVES_PER_WG + w) * ROWS_PER_WAVE;
    const int d0   = lane * 4;                  // 32 lanes * 4 = 128 = D

    // ---- phase 1: head means of K and V for 16 timesteps, stored to LDS ----
    #pragma unroll 4
    for (int r = 0; r < ROWS_PER_WAVE; ++r) {
        size_t base = ((size_t)(t0 + r) * H_TOTAL + (size_t)g * HPG) * D_DIM + d0;
        float4 ks = make_float4(0.f, 0.f, 0.f, 0.f);
        float4 vs = make_float4(0.f, 0.f, 0.f, 0.f);
        #pragma unroll
        for (int h = 0; h < HPG; ++h) {
            float4 k4 = *(const float4*)(Kc + base + (size_t)h * D_DIM);
            float4 v4 = *(const float4*)(Vc + base + (size_t)h * D_DIM);
            ks.x += k4.x; ks.y += k4.y; ks.z += k4.z; ks.w += k4.w;
            vs.x += v4.x; vs.y += v4.y; vs.z += v4.z; vs.w += v4.w;
        }
        kg[w][r][d0+0] = ks.x * 0.25f; kg[w][r][d0+1] = ks.y * 0.25f;
        kg[w][r][d0+2] = ks.z * 0.25f; kg[w][r][d0+3] = ks.w * 0.25f;
        vg[w][r][d0+0] = vs.x * 0.25f; vg[w][r][d0+1] = vs.y * 0.25f;
        vg[w][r][d0+2] = vs.z * 0.25f; vg[w][r][d0+3] = vs.w * 0.25f;
    }
    __syncthreads();

    // ---- phase 2: build B fragments (Kg^T, 32x16 bf16, 4 chunks of K=32) ---
    const int n    = lane & 15;                 // column (timestep within tile)
    const int koff = (lane < 16) ? 0 : 8;       // C-row offset for this half
    v16bf bfrag[4];
    #pragma unroll
    for (int c = 0; c < 4; ++c) {
        #pragma unroll
        for (int j = 0; j < 16; ++j)
            bfrag[c][j] = f32_to_bf16_rne(kg[w][n][c * 32 + frag_k(j, lane)]);
    }

    // ---- phase 3: scores vs 512 anchors via v_wmma_f32_16x16x32_bf16 -------
    // Double-buffered pre-packed A fragments; branch-free key-based argmax.
    const __bf16* aBase = wsA + (size_t)g * MT_CNT * FRAG_SZ;
    v16bf abuf[2][4];
    #pragma unroll
    for (int c = 0; c < 4; ++c)
        abuf[0][c] = *(const v16bf*)(aBase + ((size_t)c * 32 + lane) * 16);

    uint64_t bestkey = 0ull;
    #pragma unroll 2
    for (int mt = 0; mt < MT_CNT; ++mt) {
        const int cur = mt & 1;
        // prefetch next tile's fragments (wraps to tile 0 on last iteration)
        const __bf16* aNext = aBase + (size_t)((mt + 1) & (MT_CNT - 1)) * FRAG_SZ;
        #pragma unroll
        for (int c = 0; c < 4; ++c)
            abuf[cur ^ 1][c] = *(const v16bf*)(aNext + ((size_t)c * 32 + lane) * 16);

        v8f acc = {0.f, 0.f, 0.f, 0.f, 0.f, 0.f, 0.f, 0.f};
        #pragma unroll
        for (int c = 0; c < 4; ++c)
            acc = __builtin_amdgcn_wmma_f32_16x16x32_bf16(
                      false, abuf[cur][c], false, bfrag[c], (short)0, acc,
                      false, false);

        // per-lane scan over the 8 M-rows this lane holds (branch-free cndmask)
        float lmax = acc[0]; int lr = 0;
        #pragma unroll
        for (int r = 1; r < 8; ++r)
            if (acc[r] > lmax) { lmax = acc[r]; lr = r; }
        int lm = mt * 16 + lr + koff;           // C rows: lanes>=16 hold M=r+8

        // sortable key: higher score wins, ties -> lower anchor index
        uint64_t key = ((uint64_t)f32_orderable(lmax) << 32)
                     | (uint32_t)(M_ANCH - 1 - lm);
        // merge the two half-tiles covering the same column t (lane ^ 16)
        uint64_t okey = __shfl_xor((unsigned long long)key, 16, 32);
        key = (okey > key) ? okey : key;
        bestkey = (key > bestkey) ? key : bestkey;
    }
    int bestm = M_ANCH - 1 - (int)(bestkey & 0xFFFFFFFFu);

    // ---- phase 4: scatter-add Kg/Vg rows + counts into (g, m) accumulators -
    #pragma unroll 4
    for (int r = 0; r < ROWS_PER_WAVE; ++r) {
        int m = __shfl(bestm, r, 32);           // lane r holds column r's winner
        size_t o = ((size_t)g * M_ANCH + m) * D_DIM + d0;
        atomicAdd(wsK + o + 0, kg[w][r][d0+0]);
        atomicAdd(wsK + o + 1, kg[w][r][d0+1]);
        atomicAdd(wsK + o + 2, kg[w][r][d0+2]);
        atomicAdd(wsK + o + 3, kg[w][r][d0+3]);
        atomicAdd(wsV + o + 0, vg[w][r][d0+0]);
        atomicAdd(wsV + o + 1, vg[w][r][d0+1]);
        atomicAdd(wsV + o + 2, vg[w][r][d0+2]);
        atomicAdd(wsV + o + 3, vg[w][r][d0+3]);
        if (lane == 0) atomicAdd(wsC + (size_t)g * M_ANCH + m, 1.0f);
    }
}

// --------------------------------------- normalize + broadcast to heads ---
__global__ void k_finalize(const float* __restrict__ wsK, const float* __restrict__ wsV,
                           const float* __restrict__ wsC, float* __restrict__ out)
{
    const int MHD = M_ANCH * H_TOTAL * D_DIM;
    int i = blockIdx.x * blockDim.x + threadIdx.x;
    if (i >= MHD) return;
    int d = i % D_DIM;
    int h = (i / D_DIM) % H_TOTAL;
    int m = i / (D_DIM * H_TOTAL);
    int g = h / HPG;
    size_t a = ((size_t)g * M_ANCH + m) * D_DIM + d;
    float z = fmaxf(wsC[g * M_ANCH + m], 1e-8f);
    out[i]       = wsK[a] / z;   // K_mem (M,H,D)
    out[MHD + i] = wsV[a] / z;   // V_mem (M,H,D)
}

extern "C" void kernel_launch(void* const* d_in, const int* in_sizes, int n_in,
                              void* d_out, int out_size, void* d_ws, size_t ws_size,
                              hipStream_t stream) {
    const float* Kc      = (const float*)d_in[0];
    const float* Vc      = (const float*)d_in[1];
    const float* anchors = (const float*)d_in[2];
    float* out = (float*)d_out;

    float* wsK = (float*)d_ws;                                    // G*M*D f32
    float* wsV = wsK + (size_t)G_GRP * M_ANCH * D_DIM;            // G*M*D f32
    float* wsC = wsV + (size_t)G_GRP * M_ANCH * D_DIM;            // G*M   f32
    __bf16* wsA = (__bf16*)(wsC + (size_t)G_GRP * M_ANCH);        // G*M*D bf16

    int nacc = 2 * G_GRP * M_ANCH * D_DIM + G_GRP * M_ANCH;
    hipLaunchKernelGGL(k_init, dim3((nacc + 255) / 256), dim3(256), 0, stream,
                       wsK, nacc);

    int nfrag = G_GRP * MT_CNT * 4 * 32;  // one thread per 16-elem fragment
    hipLaunchKernelGGL(k_pack_anchors, dim3((nfrag + 255) / 256), dim3(256), 0,
                       stream, anchors, wsA);

    dim3 grid(T_TOTAL / (WAVES_PER_WG * ROWS_PER_WAVE), G_GRP);
    hipLaunchKernelGGL(k_route_accum, grid, dim3(WAVES_PER_WG * 32), 0, stream,
                       Kc, Vc, wsA, wsK, wsV, wsC);

    const int MHD = M_ANCH * H_TOTAL * D_DIM;
    hipLaunchKernelGGL(k_finalize, dim3((MHD + 255) / 256), dim3(256), 0, stream,
                       wsK, wsV, wsC, out);
}
